// TransformerDecoderBlock_33097017983496
// MI455X (gfx1250) — matrix operand order repaired and checked by reading, not compiled, not run on hardware
//
#include <hip/hip_runtime.h>
#include <hip/hip_bf16.h>

// ---------------------------------------------------------------------------
// Types for CDNA5 WMMA (wave32): v16h A/B fragments, v8f accumulator.
// ---------------------------------------------------------------------------
typedef _Float16 v8h  __attribute__((ext_vector_type(8)));
typedef _Float16 v16h __attribute__((ext_vector_type(16)));
typedef float    v8f  __attribute__((ext_vector_type(8)));
typedef unsigned int v4u __attribute__((ext_vector_type(4)));
typedef int      v8i __attribute__((ext_vector_type(8)));
typedef int      v4i __attribute__((ext_vector_type(4)));

// Pack two contiguous 8xf16 runs into a 16xf16 fragment.
// A-fragment (16x32 f16): lane = half*16 + m ; runs at K = half*8 and 16+half*8.
// B-fragment (32x16 f16): lane = half*16 + n ; K in elements, same order as A.
__device__ __forceinline__ v16h make_frag(const _Float16* lo, const _Float16* hi) {
  v8h a = *(const v8h*)lo;
  v8h b = *(const v8h*)hi;
  v16h r;
#pragma unroll
  for (int i = 0; i < 8; ++i) { r[i] = a[i]; r[i + 8] = b[i]; }
  return r;
}

// ---------------------------------------------------------------------------
// Tensor Data Mover: 2-D tile load global->LDS with LDS row padding.
// Descriptor packing per CDNA5 ISA (D# groups 0/1).
// pad codes: row of (1<<(pi+1)) dwords gets (pa+1) dwords of pad appended.
// Toolchain uses the 6-arg builtin: (v4u g0, v8i g1, v4i g2, v4i g3, v8i, cpol)
// ---------------------------------------------------------------------------
__device__ __forceinline__ void tdm_load_2d(
    unsigned lds_addr, const void* gptr,
    unsigned tile_d0, unsigned tile_d1,          // elements
    unsigned tensor_d0, unsigned tensor_d1,      // elements (OOB bound)
    unsigned long long stride0,                  // row stride, elements
    unsigned pad_interval_code, unsigned pad_amount_code) {
  unsigned long long ga = (unsigned long long)(uintptr_t)gptr;
  v4u g0;
  g0[0] = 1u;                                    // count=1, user mode, no gather
  g0[1] = lds_addr;                              // LDS byte address
  g0[2] = (unsigned)(ga & 0xFFFFFFFFu);          // global_addr[31:0]
  g0[3] = (unsigned)((ga >> 32) & 0x01FFFFFFu) | (2u << 30);  // addr[56:32] | type=2
  v8i g1;
  g1[0] = (int)((1u << 16)                       // data_size = 2 bytes
                | (1u << 20)                     // pad_enable
                | (pad_interval_code << 22)
                | (pad_amount_code << 25));
  g1[1] = (int)((tensor_d0 & 0xFFFFu) << 16);                       // dim0[15:0]
  g1[2] = (int)(((tensor_d0 >> 16) & 0xFFFFu) | ((tensor_d1 & 0xFFFFu) << 16));
  g1[3] = (int)(((tensor_d1 >> 16) & 0xFFFFu) | (tile_d0 << 16));
  g1[4] = (int)(tile_d1 & 0xFFFFu);              // tile_dim1, tile_dim2=0
  g1[5] = (int)(stride0 & 0xFFFFFFFFu);
  g1[6] = (int)((stride0 >> 32) & 0xFFFFu);      // dim1_stride unused (2D)
  g1[7] = 0;
  v4i z4 = {0, 0, 0, 0};
  v8i z8 = {0, 0, 0, 0, 0, 0, 0, 0};
  __builtin_amdgcn_tensor_load_to_lds(g0, g1, z4, z4, z8, 0);
}

// ---------------------------------------------------------------------------
// f32 -> f16 conversion (weights)
// ---------------------------------------------------------------------------
__global__ void f32_to_f16_kernel(const float* __restrict__ in,
                                  _Float16* __restrict__ out, int n) {
  int i = blockIdx.x * blockDim.x + threadIdx.x;
  if (i < n) out[i] = (_Float16)in[i];
}

// ---------------------------------------------------------------------------
// LayerNorm over D=1024, one workgroup per token, output f16.
// ---------------------------------------------------------------------------
__global__ __launch_bounds__(256) void ln_f16_kernel(
    const float* __restrict__ x, const float* __restrict__ g,
    const float* __restrict__ b, _Float16* __restrict__ out) {
  const int row = blockIdx.x;
  const float* xr = x + (size_t)row * 1024;
  float v[4], s = 0.f, ss = 0.f;
#pragma unroll
  for (int i = 0; i < 4; ++i) {
    v[i] = xr[threadIdx.x + i * 256];
    s += v[i];
    ss += v[i] * v[i];
  }
#pragma unroll
  for (int m = 16; m >= 1; m >>= 1) {
    s  += __shfl_xor(s, m, 32);
    ss += __shfl_xor(ss, m, 32);
  }
  __shared__ float ws[8], wss[8];
  const int wave = threadIdx.x >> 5, lane = threadIdx.x & 31;
  if (lane == 0) { ws[wave] = s; wss[wave] = ss; }
  __syncthreads();
  float S = 0.f, SS = 0.f;
#pragma unroll
  for (int i = 0; i < 8; ++i) { S += ws[i]; SS += wss[i]; }
  const float mu = S * (1.0f / 1024.0f);
  const float var = SS * (1.0f / 1024.0f) - mu * mu;
  const float rs = rsqrtf(var + 1e-5f);
#pragma unroll
  for (int i = 0; i < 4; ++i) {
    const int c = threadIdx.x + i * 256;
    out[(size_t)row * 1024 + c] = (_Float16)((v[i] - mu) * rs * g[c] + b[c]);
  }
}

// ---------------------------------------------------------------------------
// Tiled WMMA GEMM: C[MxN] = A[MxK](f16) @ B[KxN](f16) [+epilogue]
//   block tile 128x128, 256 threads = 8 waves (4x2), wave tile 32x64.
//   K step 32. A/B tiles DMA'd into double-buffered LDS by the TDM
//   (tensor_load_to_lds, wave 0 issues, s_wait_tensorcnt + barrier sync).
// epi: 0 = store f16 ; 1 = +bias +resid, store f32 ; 2 = +bias, GELU, store f16
// ---------------------------------------------------------------------------
#define BM 128
#define BN 128
#define BK 32
#define AS_S 48    // A LDS row stride: 16 dwords data + 8 dwords pad (96B)
#define BS_S 144   // B LDS row stride: 64 dwords data + 8 dwords pad (288B)

__global__ __launch_bounds__(256) void gemm_wmma_f16(
    const _Float16* __restrict__ A, const _Float16* __restrict__ Bm,
    void* __restrict__ out, const float* __restrict__ bias,
    const float* __restrict__ resid, int M, int N, int K, int epi) {
  __shared__ _Float16 As[2][BM * AS_S];
  __shared__ _Float16 Bs[2][BK * BS_S];
  const int tid = threadIdx.x;
  const int lane = tid & 31;
  const int wave = tid >> 5;
  const int wm = wave >> 1;  // 0..3
  const int wn = wave & 1;   // 0..1
  const int rowBlk = blockIdx.y * BM;
  const int colBlk = blockIdx.x * BN;
  const int half = lane >> 4, l16 = lane & 15;

  v8f acc[2][4];
#pragma unroll
  for (int i = 0; i < 2; ++i)
#pragma unroll
    for (int j = 0; j < 4; ++j)
#pragma unroll
      for (int e = 0; e < 8; ++e) acc[i][j][e] = 0.0f;

  const int nsteps = K / BK;
  auto issue = [&](int t, int buf) {
    const int k0 = t * BK;
    // A tile: BM x BK, row stride K ; pad 16dw rows by 8dw -> stride 48 f16
    tdm_load_2d((unsigned)(uintptr_t)&As[buf][0],
                A + (size_t)rowBlk * K + k0,
                BK, BM, (unsigned)(K - k0), (unsigned)(M - rowBlk),
                (unsigned long long)K, 3u, 7u);
    // B tile: BK x BN, row stride N ; pad 64dw rows by 8dw -> stride 144 f16
    tdm_load_2d((unsigned)(uintptr_t)&Bs[buf][0],
                Bm + (size_t)k0 * N + colBlk,
                BN, BK, (unsigned)(N - colBlk), (unsigned)(K - k0),
                (unsigned long long)N, 5u, 7u);
  };

  if (wave == 0) issue(0, 0);
  int buf = 0;
  for (int t = 0; t < nsteps; ++t) {
    if (wave == 0) {
      if (t + 1 < nsteps) {
        issue(t + 1, buf ^ 1);                       // prefetch next tile pair
        __builtin_amdgcn_s_wait_tensorcnt(2);        // current pair complete
      } else {
        __builtin_amdgcn_s_wait_tensorcnt(0);
      }
    }
    __syncthreads();

    v16h afrag[2], bfrag[4];
#pragma unroll
    for (int mt = 0; mt < 2; ++mt) {
      const int r = wm * 32 + mt * 16 + l16;
      afrag[mt] = make_frag(&As[buf][r * AS_S + half * 8],
                            &As[buf][r * AS_S + 16 + half * 8]);
    }
#pragma unroll
    for (int nt = 0; nt < 4; ++nt) {
      const int c0 = wn * 64 + nt * 16;
#pragma unroll
      for (int s = 0; s < 16; ++s)
        bfrag[nt][s] = Bs[buf][((s < 8) ? (half * 8 + s) : (16 + half * 8 + s - 8)) * BS_S + c0 + l16];
    }
#pragma unroll
    for (int mt = 0; mt < 2; ++mt)
#pragma unroll
      for (int nt = 0; nt < 4; ++nt)
        acc[mt][nt] = __builtin_amdgcn_wmma_f32_16x16x32_f16(
            false, afrag[mt], false, bfrag[nt], (short)0, acc[mt][nt],
            false, false);
    __syncthreads();
    buf ^= 1;
  }

  // Epilogue: C layout -> VGPR r : M = r + 8*half, N = l16
#pragma unroll
  for (int mt = 0; mt < 2; ++mt) {
#pragma unroll
    for (int nt = 0; nt < 4; ++nt) {
      const int col = colBlk + wn * 64 + nt * 16 + l16;
      const int row0 = rowBlk + wm * 32 + mt * 16 + half * 8;
      const float bv = bias ? bias[col] : 0.0f;
#pragma unroll
      for (int r = 0; r < 8; ++r) {
        const size_t idx = (size_t)(row0 + r) * N + col;
        float v = acc[mt][nt][r] + bv;
        if (epi == 2) v = 0.5f * v * (1.0f + erff(v * 0.70710678118f));
        if (epi == 1)
          ((float*)out)[idx] = v + resid[idx];
        else
          ((_Float16*)out)[idx] = (_Float16)v;
      }
    }
  }
}

// ---------------------------------------------------------------------------
// Causal flash attention: grid (S/64, B*H), 128 threads = 4 waves.
// Each wave owns a 16-row Q tile. We compute S^T = K @ Q^T so that:
//   * A-fragments (K rows, contiguous d) come straight from global memory
//   * Q^T is transposed into LDS ONCE per wave (not per key-block)
//   * softmax rows land in the lane (N) dimension -> in-lane reductions +
//     a single shfl_xor(16); m/l are scalars per lane.
// P is round-tripped through LDS to A-fragment layout; V rows are read
// directly from global in exact B-fragment layout (lane = d column, keys in elements).
// ---------------------------------------------------------------------------
#define QTS 24  // Qt row stride (f16): 48B rows, 16B aligned
#define PTS 48  // P  row stride (f16): 96B rows, 16B aligned

__global__ __launch_bounds__(128) void attn_wmma_kernel(
    const _Float16* __restrict__ Q, const _Float16* __restrict__ Kmat,
    const _Float16* __restrict__ V, _Float16* __restrict__ CTX) {
  __shared__ _Float16 QtAll[4 * 64 * QTS];  // per-wave Q^T: 64 d x 16 q
  __shared__ _Float16 PAll[4 * 16 * PTS];   // per-wave P: 16 q x 32 keys
  const int wave = threadIdx.x >> 5;
  const int lane = threadIdx.x & 31;
  const int half = lane >> 4, l16 = lane & 15;
  const int bh = blockIdx.y;  // b*16 + h
  const int b = bh >> 4, h = bh & 15;
  const int q0 = blockIdx.x * 64 + wave * 16;
  const size_t tokBase = (size_t)b * 2048;
  _Float16* Qt = &QtAll[wave * 64 * QTS];
  _Float16* P  = &PAll[wave * 16 * PTS];

  // Stage Q^T once: lane covers q = l16, d in [half*32, half*32+32)
  {
    const _Float16* qrow = Q + (tokBase + q0 + l16) * 1024 + h * 64 + half * 32;
#pragma unroll
    for (int c8 = 0; c8 < 4; ++c8) {
      v8h qv = *(const v8h*)(qrow + c8 * 8);
#pragma unroll
      for (int j = 0; j < 8; ++j)
        Qt[(half * 32 + c8 * 8 + j) * QTS + l16] = qv[j];
    }
  }
  // Q^T B-fragments (q = l16, d in table order within 32-chunk) - reused
  v16h qtf[2];
#pragma unroll
  for (int c = 0; c < 2; ++c)
#pragma unroll
    for (int s = 0; s < 16; ++s)
      qtf[c][s] = Qt[(c * 32 + ((s < 8) ? (half * 8 + s) : (16 + half * 8 + s - 8))) * QTS + l16];

  float m = -1e30f, l = 0.f;
  v8f o[4];
#pragma unroll
  for (int dt = 0; dt < 4; ++dt)
#pragma unroll
    for (int e = 0; e < 8; ++e) o[dt][e] = 0.f;

  const int nblk = (q0 + 16 + 31) >> 5;  // causal: keys <= q0+15
  for (int blk = 0; blk < nblk; ++blk) {
    const int kbase = blk * 32;

    // S^T tiles (16 keys x 16 q), kh = key half; contraction d = 64 in 2 chunks
    v8f s[2];
#pragma unroll
    for (int kh = 0; kh < 2; ++kh)
#pragma unroll
      for (int e = 0; e < 8; ++e) s[kh][e] = 0.f;
#pragma unroll
    for (int kh = 0; kh < 2; ++kh) {
      const _Float16* krow =
          Kmat + (tokBase + kbase + kh * 16 + l16) * 1024 + h * 64;
#pragma unroll
      for (int c = 0; c < 2; ++c) {
        v16h kf = make_frag(krow + c * 32 + half * 8,
                            krow + c * 32 + 16 + half * 8);
        s[kh] = __builtin_amdgcn_wmma_f32_16x16x32_f16(
            false, kf, false, qtf[c], (short)0, s[kh], false, false);
      }
    }

    // Per-lane query column q = q0 + l16 ; keys live in (kh, r, half)
    const int qrow = q0 + l16;
    float sv[2][8];
    float mx = -1e30f;
#pragma unroll
    for (int kh = 0; kh < 2; ++kh)
#pragma unroll
      for (int r = 0; r < 8; ++r) {
        const int key = kbase + kh * 16 + r + 8 * half;
        float val = s[kh][r] * 0.125f;  // 1/sqrt(64)
        if (key > qrow) val = -1e9f;
        sv[kh][r] = val;
        mx = fmaxf(mx, val);
      }
    mx = fmaxf(mx, __shfl_xor(mx, 16, 32));
    const float mn = fmaxf(m, mx);
    const float sc = __expf(m - mn);
    float rs = 0.f;
#pragma unroll
    for (int kh = 0; kh < 2; ++kh)
#pragma unroll
      for (int r = 0; r < 8; ++r) {
        const float p = __expf(sv[kh][r] - mn);
        sv[kh][r] = p;
        rs += p;
      }
    rs += __shfl_xor(rs, 16, 32);
    l = l * sc + rs;
    m = mn;
    // Rescale O (O layout: q = r + 8*half) - redistribute sc across lanes
#pragma unroll
    for (int r = 0; r < 8; ++r) {
      const float scr = __shfl(sc, r + 8 * half, 32);
#pragma unroll
      for (int dt = 0; dt < 4; ++dt) o[dt][r] *= scr;
    }

    // P -> LDS in row-major (q x 32 keys), reload as A-fragment
#pragma unroll
    for (int kh = 0; kh < 2; ++kh)
#pragma unroll
      for (int r = 0; r < 8; ++r)
        P[l16 * PTS + kh * 16 + r + 8 * half] = (_Float16)sv[kh][r];
    const v16h pf = make_frag(&P[l16 * PTS + half * 8],
                              &P[l16 * PTS + 16 + half * 8]);

    // O += P V : V from global as B-fragments (n = l16, keys in table order)
    const _Float16* vrow = V + (tokBase + kbase) * 1024 + h * 64;
#pragma unroll
    for (int dt = 0; dt < 4; ++dt) {
      v16h vf;
#pragma unroll
      for (int s = 0; s < 16; ++s)
        vf[s] = vrow[(size_t)((s < 8) ? (half * 8 + s) : (16 + half * 8 + s - 8)) * 1024 + dt * 16 + l16];
      o[dt] = __builtin_amdgcn_wmma_f32_16x16x32_f16(
          false, pf, false, vf, (short)0, o[dt], false, false);
    }
  }

  // Finalize: ctx = O / l ; redistribute l into O's row layout
  float lr[8];
#pragma unroll
  for (int r = 0; r < 8; ++r) lr[r] = __shfl(l, r + 8 * half, 32);
#pragma unroll
  for (int dt = 0; dt < 4; ++dt) {
    const int col = h * 64 + dt * 16 + l16;
#pragma unroll
    for (int r = 0; r < 8; ++r) {
      const size_t t = tokBase + q0 + r + 8 * half;
      CTX[t * 1024 + col] = (_Float16)(o[dt][r] / lr[r]);
    }
  }
}

// ---------------------------------------------------------------------------
// Orchestration
// ---------------------------------------------------------------------------
extern "C" void kernel_launch(void* const* d_in, const int* in_sizes, int n_in,
                              void* d_out, int out_size, void* d_ws,
                              size_t ws_size, hipStream_t stream) {
  (void)in_sizes; (void)n_in; (void)out_size; (void)ws_size;
  const float* x     = (const float*)d_in[0];
  const float* w_q   = (const float*)d_in[1];
  const float* w_k   = (const float*)d_in[2];
  const float* w_v   = (const float*)d_in[3];
  const float* w_o   = (const float*)d_in[4];
  const float* b_o   = (const float*)d_in[5];
  const float* ln1_g = (const float*)d_in[6];
  const float* ln1_b = (const float*)d_in[7];
  const float* ln2_g = (const float*)d_in[8];
  const float* ln2_b = (const float*)d_in[9];
  const float* w1    = (const float*)d_in[10];
  const float* b1    = (const float*)d_in[11];
  const float* w2    = (const float*)d_in[12];
  const float* b2    = (const float*)d_in[13];

  const int MT = 4096;  // B*S tokens
  const int D = 1024, FF = 4096;

  char* ws = (char*)d_ws;
  size_t off = 0;
  auto alloc = [&](size_t bytes) -> void* {
    void* p = ws + off;
    off += (bytes + 255) & ~(size_t)255;
    return p;
  };
  _Float16* wq16 = (_Float16*)alloc((size_t)D * D * 2);
  _Float16* wk16 = (_Float16*)alloc((size_t)D * D * 2);
  _Float16* wv16 = (_Float16*)alloc((size_t)D * D * 2);
  _Float16* wo16 = (_Float16*)alloc((size_t)D * D * 2);
  _Float16* w116 = (_Float16*)alloc((size_t)D * FF * 2);
  _Float16* w216 = (_Float16*)alloc((size_t)FF * D * 2);
  _Float16* h16  = (_Float16*)alloc((size_t)MT * D * 2);  // ln1 out, reused as ctx
  _Float16* q16  = (_Float16*)alloc((size_t)MT * D * 2);  // Q, reused as ln2 out
  _Float16* k16  = (_Float16*)alloc((size_t)MT * D * 2);
  _Float16* v16  = (_Float16*)alloc((size_t)MT * D * 2);
  float*    x1   = (float*)alloc((size_t)MT * D * 4);
  _Float16* ff1  = (_Float16*)alloc((size_t)MT * FF * 2);
  _Float16* ctx16 = h16;  // h16 dead after QKV GEMMs
  _Float16* h2_16 = q16;  // q16 dead after attention

  // 1) weights f32 -> f16
  const int CT = 256;
  f32_to_f16_kernel<<<(D * D + CT - 1) / CT, CT, 0, stream>>>(w_q, wq16, D * D);
  f32_to_f16_kernel<<<(D * D + CT - 1) / CT, CT, 0, stream>>>(w_k, wk16, D * D);
  f32_to_f16_kernel<<<(D * D + CT - 1) / CT, CT, 0, stream>>>(w_v, wv16, D * D);
  f32_to_f16_kernel<<<(D * D + CT - 1) / CT, CT, 0, stream>>>(w_o, wo16, D * D);
  f32_to_f16_kernel<<<(D * FF + CT - 1) / CT, CT, 0, stream>>>(w1, w116, D * FF);
  f32_to_f16_kernel<<<(FF * D + CT - 1) / CT, CT, 0, stream>>>(w2, w216, FF * D);

  // 2) ln1(x) -> h16
  ln_f16_kernel<<<MT, 256, 0, stream>>>(x, ln1_g, ln1_b, h16);

  // 3) QKV projections (no bias), f16 outputs
  dim3 gD(D / BN, MT / BM);  // (8, 32)
  gemm_wmma_f16<<<gD, 256, 0, stream>>>(h16, wq16, q16, nullptr, nullptr, MT, D, D, 0);
  gemm_wmma_f16<<<gD, 256, 0, stream>>>(h16, wk16, k16, nullptr, nullptr, MT, D, D, 0);
  gemm_wmma_f16<<<gD, 256, 0, stream>>>(h16, wv16, v16, nullptr, nullptr, MT, D, D, 0);

  // 4) causal flash attention -> ctx16
  attn_wmma_kernel<<<dim3(2048 / 64, 32), 128, 0, stream>>>(q16, k16, v16, ctx16);

  // 5) output projection + bias + residual(x) -> x1 (f32)
  gemm_wmma_f16<<<gD, 256, 0, stream>>>(ctx16, wo16, x1, b_o, x, MT, D, D, 1);

  // 6) ln2(x1) -> h2_16
  ln_f16_kernel<<<MT, 256, 0, stream>>>(x1, ln2_g, ln2_b, h2_16);

  // 7) FFN1: bias + exact GELU -> ff1 (f16)
  gemm_wmma_f16<<<dim3(FF / BN, MT / BM), 256, 0, stream>>>(
      h2_16, w116, ff1, b1, nullptr, MT, FF, D, 2);

  // 8) FFN2: bias + residual(x1) -> d_out (f32)
  gemm_wmma_f16<<<gD, 256, 0, stream>>>(ff1, w216, (float*)d_out, b2, x1,
                                        MT, D, FF, 1);
}
